// EncoderNet_63763084476775
// MI455X (gfx1250) — compile-verified, matrix operand order checked by reading, not run on hardware
//
#include <hip/hip_runtime.h>
#include <stddef.h>

// ---------------------------------------------------------------------------
// Types / helpers
// ---------------------------------------------------------------------------
typedef __attribute__((ext_vector_type(16))) __bf16 v16bf;
typedef __attribute__((ext_vector_type(8)))  float  v8f;

union Frag {                       // one 16-bit WMMA A/B fragment per lane
    v16bf v;
    uint4 q[2];                    // two 16-byte halves
    unsigned short s[16];
};

__device__ __forceinline__ unsigned short f2bf(float f) {
    unsigned int u = __float_as_uint(f);
    u += 0x7FFFu + ((u >> 16) & 1u);   // round-to-nearest-even
    return (unsigned short)(u >> 16);
}
__device__ __forceinline__ float bf2f(unsigned short h) {
    return __uint_as_float(((unsigned int)h) << 16);
}

// K index inside a 32-wide chunk for 16-bit WMMA fragments (ISA 7.12.2):
// lanes 0-15 hold K {0..7, 16..23}; lanes 16-31 hold K {8..15, 24..31}.
__device__ __forceinline__ int frag_k(int e, int half) {
    return e + ((e >= 8) ? 8 : 0) + half * 8;
}

// ---------------------------------------------------------------------------
// Weight packing: conv OIHW fp32 -> bf16 fragment order
// layout: Wp[t][ct][kc][lane][e],  t = ky*3+kx
// ---------------------------------------------------------------------------
__global__ __launch_bounds__(256)
void pack_conv_w(const float* __restrict__ W, unsigned short* __restrict__ Wp,
                 int Cout, int Cin)
{
    const int KC = (Cin + 31) >> 5;
    const int ctiles = Cout >> 4;
    const long total = 9l * ctiles * KC * 512;
    const long idx = (long)blockIdx.x * 256 + threadIdx.x;
    if (idx >= total) return;
    const int e    = (int)(idx & 15);
    const int lane = (int)((idx >> 4) & 31);
    long rest = idx >> 9;
    const int kc = (int)(rest % KC); rest /= KC;
    const int ct = (int)(rest % ctiles);
    const int t  = (int)(rest / ctiles);
    const int ky = t / 3, kx = t - ky * 3;
    const int cin  = kc * 32 + frag_k(e, lane >> 4);
    const int cout = ct * 16 + (lane & 15);
    const float v = (cin < Cin) ? W[(((size_t)cout * Cin + cin) * 3 + ky) * 3 + kx] : 0.f;
    Wp[idx] = f2bf(v);
}

// ---------------------------------------------------------------------------
// Weight packing: GEMM [K,N] fp32 -> bf16 fragment order, optional row perm
// (folds the reference's NCHW flatten into NHWC order:
//  row = (k % permC) * permP + k / permC when permP > 0)
// layout: Bp[nt][kc][lane][e]
// ---------------------------------------------------------------------------
__global__ __launch_bounds__(256)
void pack_gemm_w(const float* __restrict__ W, unsigned short* __restrict__ Bp,
                 int K, int N, int permP, int permC)
{
    const int KC = K >> 5;
    const long total = (long)(N >> 4) * KC * 512;
    const long idx = (long)blockIdx.x * 256 + threadIdx.x;
    if (idx >= total) return;
    const int e    = (int)(idx & 15);
    const int lane = (int)((idx >> 4) & 31);
    long rest = idx >> 9;
    const int kc = (int)(rest % KC);
    const int nt = (int)(rest / KC);
    const int k  = kc * 32 + frag_k(e, lane >> 4);
    int row = k;
    if (permP > 0) row = (k % permC) * permP + (k / permC);
    const int col = nt * 16 + (lane & 15);
    Bp[idx] = f2bf(W[(size_t)row * N + col]);
}

// ---------------------------------------------------------------------------
// Front end -> att in padded NHWC bf16: att[b][m][n][c], c-stride 32 (12 real)
// ---------------------------------------------------------------------------
__global__ __launch_bounds__(256)
void frontend_att(const float* __restrict__ x, const float* __restrict__ loc,
                  const int* __restrict__ rand_order,
                  const float* __restrict__ Wd, const float* __restrict__ bd,
                  const float* __restrict__ Wl, const float* __restrict__ bl,
                  const float* __restrict__ We, const float* __restrict__ be,
                  const float* __restrict__ av,
                  unsigned short* __restrict__ att)
{
    __shared__ float emb[69][60];
    __shared__ float w1s[6][69];
    __shared__ float w2s[6][69];

    const int b = blockIdx.x >> 1;
    const int i = blockIdx.x & 1;
    const int* order = rand_order + i * 62;
    const int t = threadIdx.x;

    for (int idx = t; idx < 69 * 60; idx += 256) {
        const int m = idx / 60, j = idx - m * 60;
        float v;
        if (m < 62) {
            const int node = order[m];
            const float* xg = x + b * 380 + node * 5;
            float fe = bd[j];
            #pragma unroll
            for (int f = 0; f < 5; ++f) fe += xg[f] * Wd[f * 60 + j];
            const float* lv = loc + node * 3;
            float le = bl[j];
            #pragma unroll
            for (int f = 0; f < 3; ++f) le += lv[f] * Wl[f * 60 + j];
            v = fe + le;
            v = v > 0.f ? v : 0.f;
        } else {
            const int s = m - 62;
            const float* xe = x + b * 380 + 310 + s * 10;
            float ev = be[j];
            #pragma unroll
            for (int f = 0; f < 10; ++f) ev += xe[f] * We[f * 60 + j];
            v = ev > 0.f ? ev : 0.f;
        }
        emb[m][j] = v;
    }
    __syncthreads();

    for (int idx = t; idx < 6 * 69; idx += 256) {
        const int h = idx / 69, m = idx - h * 69;
        float s1 = 0.f, s2 = 0.f;
        #pragma unroll
        for (int d = 0; d < 10; ++d) {
            const float e = emb[m][h * 10 + d];
            s1 += e * av[d];
            s2 += e * av[10 + d];
        }
        w1s[h][m] = s1;
        w2s[h][m] = s2;
    }
    __syncthreads();

    unsigned short* ab = att + (size_t)b * 4761 * 32;
    for (int idx = t; idx < 6 * 69 * 69; idx += 256) {
        const int h = idx / (69 * 69);
        const int r = idx - h * 69 * 69;          // r = m*69 + n
        ab[(size_t)r * 32 + i * 6 + h] = f2bf(w1s[h][r / 69] + w2s[h][r % 69]);
    }
    if (i == 0) {  // zero the 20 padded channels
        for (int idx = t; idx < 4761 * 20; idx += 256) {
            ab[(size_t)(idx / 20) * 32 + 12 + (idx % 20)] = 0;
        }
    }
}

// ---------------------------------------------------------------------------
// Per-channel BN stats over NHWC bf16 (channel stride Cs) -> (scale, shift)
// ---------------------------------------------------------------------------
__global__ __launch_bounds__(256)
void bn_stats_nhwc(const unsigned short* __restrict__ Y,
                   const float* __restrict__ gamma, const float* __restrict__ beta,
                   float* __restrict__ scsh, int Cs, long NP)
{
    __shared__ float ssum[256];
    __shared__ float ssq[256];
    const int c = blockIdx.x, t = threadIdx.x;
    float s = 0.f, q = 0.f;
    for (long i = t; i < NP; i += 256) {
        const float v = bf2f(Y[i * Cs + c]);
        s += v; q += v * v;
    }
    ssum[t] = s; ssq[t] = q;
    __syncthreads();
    for (int off = 128; off > 0; off >>= 1) {
        if (t < off) { ssum[t] += ssum[t + off]; ssq[t] += ssq[t + off]; }
        __syncthreads();
    }
    if (t == 0) {
        const float inv  = 1.0f / (float)NP;
        const float m    = ssum[0] * inv;
        const float var  = ssq[0] * inv - m * m;
        const float rstd = rsqrtf(var + 1e-5f);
        const float sc   = gamma[c] * rstd;
        scsh[2 * c]     = sc;
        scsh[2 * c + 1] = beta[c] - m * sc;
    }
}

// ---------------------------------------------------------------------------
// In-place per-channel affine on NHWC bf16 (BN of the att tensor)
// ---------------------------------------------------------------------------
__global__ __launch_bounds__(256)
void bn_apply_nhwc(unsigned short* __restrict__ Y, const float* __restrict__ scsh,
                   int C, int Cs, long NP)
{
    const long total = NP * C;
    for (long idx = (long)blockIdx.x * 256 + threadIdx.x; idx < total;
         idx += (long)gridDim.x * 256) {
        const int c = (int)(idx % C);
        const long p = idx / C;
        unsigned short* q = Y + p * Cs + c;
        *q = f2bf(bf2f(*q) * scsh[2 * c] + scsh[2 * c + 1]);
    }
}

// ---------------------------------------------------------------------------
// Implicit-GEMM 3x3 conv, NHWC bf16 in/out, packed weights, fp32 WMMA acc.
// Each wave: 64 output channels x 16 x-positions, 9-tap loop, K = Cin chunks.
// All fragment traffic is b128 loads; output is one b128 store per lane/tile.
// One wave per block -> EXEC all ones at the WMMA (ISA requirement).
// ---------------------------------------------------------------------------
__global__ __launch_bounds__(32)
void conv_wmma_nhwc(const unsigned short* __restrict__ X,
                    const unsigned short* __restrict__ Wp,
                    unsigned short* __restrict__ Y,
                    int B, int Cin, int CinS, int Hin, int Win,
                    int Cout, int Hout, int Wout, int pad)
{
    const int lane = threadIdx.x;
    const int lr = lane & 15, half = lane >> 4;
    const int x0 = blockIdx.x * 16;
    const int y  = blockIdx.y;
    const int c64 = Cout >> 6;
    const int cb  = (blockIdx.z % c64) * 4;        // base 16-wide cout tile
    const int b   = blockIdx.z / c64;
    const int KC = (Cin + 31) >> 5;
    const int ctiles = Cout >> 4;

    v8f acc[4];
    const v8f vz = {0.f, 0.f, 0.f, 0.f, 0.f, 0.f, 0.f, 0.f};
    #pragma unroll
    for (int i = 0; i < 4; ++i) acc[i] = vz;

    const int xin = x0 + lr - pad;

    for (int t = 0; t < 9; ++t) {
        const int ky = t / 3, kx = t - ky * 3;
        const int yy = y - pad + ky;
        if (yy < 0 || yy >= Hin) continue;         // uniform skip
        const int xx = xin + kx;
        const bool xok = (xx >= 0) && (xx < Win);
        const unsigned short* xrow =
            X + ((size_t)(b * Hin + yy) * Win + xx) * CinS;
        for (int kc = 0; kc < KC; ++kc) {
            Frag bf;
            if (xok) {
                const unsigned short* p = xrow + kc * 32 + half * 8;
                bf.q[0] = *(const uint4*)(p);
                bf.q[1] = *(const uint4*)(p + 16);
            } else {
                bf.q[0] = make_uint4(0u, 0u, 0u, 0u);
                bf.q[1] = make_uint4(0u, 0u, 0u, 0u);
            }
            #pragma unroll
            for (int i = 0; i < 4; ++i) {
                Frag af;
                const unsigned short* wp =
                    Wp + ((((size_t)t * ctiles + (cb + i)) * KC + kc) * 32 + lane) * 16;
                af.q[0] = *(const uint4*)(wp);
                af.q[1] = *(const uint4*)(wp + 8);
                acc[i] = __builtin_amdgcn_wmma_f32_16x16x32_bf16(
                    false, af.v, false, bf.v, (short)0, acc[i], false, false);
            }
        }
    }

    const int xo = x0 + lr;
    if (xo < Wout) {
        unsigned short* yp = Y + ((size_t)(b * Hout + y) * Wout + xo) * Cout;
        #pragma unroll
        for (int i = 0; i < 4; ++i) {
            Frag o;
            #pragma unroll
            for (int r = 0; r < 8; ++r) o.s[r] = f2bf(acc[i][r]);
            *(uint4*)(yp + (cb + i) * 16 + half * 8) = o.q[0];
        }
    }
}

// ---------------------------------------------------------------------------
// BN affine + 2x2/2 maxpool + LeakyReLU on NHWC bf16, 8-channel b128 groups
// ---------------------------------------------------------------------------
__global__ __launch_bounds__(256)
void pool_nhwc(const unsigned short* __restrict__ Y, const float* __restrict__ scsh,
               unsigned short* __restrict__ Xo,
               int B, int C, int H, int W, int Ho, int Wo,
               float slope, int do_affine, int do_act)
{
    const int CG = C >> 3;
    const long total = (long)B * Ho * Wo * CG;
    for (long i = (long)blockIdx.x * 256 + threadIdx.x; i < total;
         i += (long)gridDim.x * 256) {
        const int cg = (int)(i % CG);
        long r = i / CG;
        const int xo = (int)(r % Wo); r /= Wo;
        const int yo = (int)(r % Ho);
        const int b  = (int)(r / Ho);
        const unsigned short* p =
            Y + ((size_t)(b * H + 2 * yo) * W + 2 * xo) * C + cg * 8;
        Frag a0, a1, a2, a3;
        a0.q[0] = *(const uint4*)(p);
        a1.q[0] = *(const uint4*)(p + C);
        a2.q[0] = *(const uint4*)(p + (size_t)W * C);
        a3.q[0] = *(const uint4*)(p + (size_t)W * C + C);
        Frag o;
        #pragma unroll
        for (int j = 0; j < 8; ++j) {
            float sc = 1.f, sh = 0.f;
            if (do_affine) {
                const int c = cg * 8 + j;
                sc = scsh[2 * c]; sh = scsh[2 * c + 1];
            }
            const float v0 = bf2f(a0.s[j]) * sc + sh;
            const float v1 = bf2f(a1.s[j]) * sc + sh;
            const float v2 = bf2f(a2.s[j]) * sc + sh;
            const float v3 = bf2f(a3.s[j]) * sc + sh;
            float m = fmaxf(fmaxf(v0, v1), fmaxf(v2, v3));
            if (do_act) m = m >= 0.f ? m : slope * m;
            o.s[j] = f2bf(m);
        }
        *(uint4*)(Xo + i * 8) = o.q[0];
    }
}

// ---------------------------------------------------------------------------
// WMMA GEMM: C[M,N] = act(A@W + bias).  A: bf16 [M][K] row-contiguous;
// W: pre-packed fragment layout.  4 N-tiles per wave (A-fragment reuse).
// ---------------------------------------------------------------------------
__global__ __launch_bounds__(32)
void gemm_wmma_packed(const unsigned short* __restrict__ A,
                      const unsigned short* __restrict__ Bp,
                      const float* __restrict__ bias, float* __restrict__ C,
                      int M, int N, int K, int act)
{
    const int lane = threadIdx.x;
    const int lr = lane & 15, half = lane >> 4;
    const int ntb = blockIdx.x * 4;
    const int m0  = blockIdx.y * 16;
    const int KC  = K >> 5;

    v8f acc[4];
    const v8f vz = {0.f, 0.f, 0.f, 0.f, 0.f, 0.f, 0.f, 0.f};
    #pragma unroll
    for (int i = 0; i < 4; ++i) acc[i] = vz;

    const unsigned short* arow = A + (size_t)(m0 + lr) * K;
    for (int kc = 0; kc < KC; ++kc) {
        Frag af;
        const unsigned short* p = arow + kc * 32 + half * 8;
        af.q[0] = *(const uint4*)(p);
        af.q[1] = *(const uint4*)(p + 16);
        #pragma unroll
        for (int i = 0; i < 4; ++i) {
            Frag bf;
            const unsigned short* wp =
                Bp + (((size_t)(ntb + i) * KC + kc) * 32 + lane) * 16;
            bf.q[0] = *(const uint4*)(wp);
            bf.q[1] = *(const uint4*)(wp + 8);
            acc[i] = __builtin_amdgcn_wmma_f32_16x16x32_bf16(
                false, af.v, false, bf.v, (short)0, acc[i], false, false);
        }
    }

    #pragma unroll
    for (int i = 0; i < 4; ++i) {
        const int n = (ntb + i) * 16 + lr;
        #pragma unroll
        for (int r = 0; r < 8; ++r) {
            const int m = r + half * 8;
            float v = acc[i][r] + bias[n];
            if (act) v = v >= 0.f ? v : 0.1f * v;
            C[(size_t)(m0 + m) * N + n] = v;
        }
    }
}

// ---------------------------------------------------------------------------
// BN1d over batch (blockDim == B == 128); optional bf16 output (for the
// concat buffer feeding the next bf16 GEMM).
// ---------------------------------------------------------------------------
__global__ __launch_bounds__(128)
void bn1d(const float* __restrict__ in, const float* __restrict__ gamma,
          const float* __restrict__ beta, void* __restrict__ out,
          int Bn, int F, int outF, int outOff, int out_bf16)
{
    __shared__ float ssum[128];
    __shared__ float ssq[128];
    __shared__ float scsh[2];
    const int f = blockIdx.x, t = threadIdx.x;
    const float v = in[(size_t)t * F + f];
    ssum[t] = v; ssq[t] = v * v;
    __syncthreads();
    for (int off = 64; off > 0; off >>= 1) {
        if (t < off) { ssum[t] += ssum[t + off]; ssq[t] += ssq[t + off]; }
        __syncthreads();
    }
    if (t == 0) {
        const float m    = ssum[0] / (float)Bn;
        const float var  = ssq[0] / (float)Bn - m * m;
        const float rstd = rsqrtf(var + 1e-5f);
        scsh[0] = gamma[f] * rstd;
        scsh[1] = beta[f] - m * gamma[f] * rstd;
    }
    __syncthreads();
    const float nv = v * scsh[0] + scsh[1];
    if (out_bf16) ((unsigned short*)out)[(size_t)t * outF + outOff + f] = f2bf(nv);
    else          ((float*)out)[(size_t)t * outF + outOff + f] = nv;
}

// ---------------------------------------------------------------------------
// Final tiny FC: out[B,5] = h[B,512] @ W1[512,5] + b1out
// ---------------------------------------------------------------------------
__global__ __launch_bounds__(128)
void final_fc(const float* __restrict__ h, const float* __restrict__ W1,
              const float* __restrict__ b1, float* __restrict__ out,
              int Bn, int K, int Ncls)
{
    const int idx = blockIdx.x * blockDim.x + threadIdx.x;
    if (idx >= Bn * Ncls) return;
    const int b = idx / Ncls, c = idx - b * Ncls;
    float s = b1[c];
    for (int k = 0; k < K; ++k) s += h[(size_t)b * K + k] * W1[k * Ncls + c];
    out[idx] = s;
}

// ---------------------------------------------------------------------------
// Host driver
// ---------------------------------------------------------------------------
extern "C" void kernel_launch(void* const* d_in, const int* in_sizes, int n_in,
                              void* d_out, int out_size, void* d_ws, size_t ws_size,
                              hipStream_t stream)
{
    (void)in_sizes; (void)n_in; (void)out_size; (void)ws_size;
    const float* x     = (const float*)d_in[0];
    const float* loc   = (const float*)d_in[1];
    const int*   ro    = (const int*)  d_in[2];
    const float* Wd    = (const float*)d_in[3];
    const float* bd    = (const float*)d_in[4];
    const float* Wl    = (const float*)d_in[5];
    const float* bl    = (const float*)d_in[6];
    const float* We    = (const float*)d_in[7];
    const float* be    = (const float*)d_in[8];
    const float* av    = (const float*)d_in[9];
    const float* g_in  = (const float*)d_in[10];
    const float* b_in  = (const float*)d_in[11];
    const float* c1    = (const float*)d_in[12];
    const float* g1    = (const float*)d_in[13];
    const float* b1    = (const float*)d_in[14];
    const float* c2    = (const float*)d_in[15];
    const float* g2    = (const float*)d_in[16];
    const float* b2    = (const float*)d_in[17];
    const float* c3    = (const float*)d_in[18];
    const float* g3    = (const float*)d_in[19];
    const float* b3    = (const float*)d_in[20];
    const float* c4    = (const float*)d_in[21];
    const float* g4    = (const float*)d_in[22];
    const float* b4    = (const float*)d_in[23];
    const float* Wlast = (const float*)d_in[24];
    const float* blast = (const float*)d_in[25];
    const float* glast = (const float*)d_in[26];
    const float* bglast= (const float*)d_in[27];
    const float* Wsec  = (const float*)d_in[28];
    const float* bsec  = (const float*)d_in[29];
    const float* gsec  = (const float*)d_in[30];
    const float* bgsec = (const float*)d_in[31];
    const float* W0    = (const float*)d_in[32];
    const float* b0    = (const float*)d_in[33];
    const float* g0    = (const float*)d_in[34];
    const float* bg0   = (const float*)d_in[35];
    const float* W1p   = (const float*)d_in[36];
    const float* b1out = (const float*)d_in[37];
    float* dout = (float*)d_out;

    const int B = 128;

    char* ws = (char*)d_ws;
    size_t off = 0;
    auto suballoc = [&](size_t bytes) -> char* {
        char* p = ws + off;
        off = (off + bytes + 255) & ~(size_t)255;
        return p;
    };

    // activations (NHWC bf16)
    unsigned short* att = (unsigned short*)suballoc((size_t)B * 4761 * 32 * 2);
    unsigned short* Y1  = (unsigned short*)suballoc((size_t)B * 4761 * 128 * 2);
    unsigned short* X2  = (unsigned short*)suballoc((size_t)B * 1156 * 128 * 2);
    unsigned short* Y2  = (unsigned short*)suballoc((size_t)B * 1024 * 192 * 2);
    unsigned short* X3  = (unsigned short*)suballoc((size_t)B * 256 * 192 * 2);
    unsigned short* Y3  = (unsigned short*)suballoc((size_t)B * 256 * 256 * 2);
    unsigned short* X4  = (unsigned short*)suballoc((size_t)B * 64 * 256 * 2);
    unsigned short* Y4  = (unsigned short*)suballoc((size_t)B * 64 * 512 * 2);
    unsigned short* X5  = (unsigned short*)suballoc((size_t)B * 16 * 512 * 2);   // f1
    unsigned short* F2  = (unsigned short*)suballoc((size_t)B * 4096 * 2);       // f2
    // packed weights (bf16 fragment order)
    unsigned short* Wp1 = (unsigned short*)suballoc(9ul * 8  * 1 * 512 * 2);
    unsigned short* Wp2 = (unsigned short*)suballoc(9ul * 12 * 4 * 512 * 2);
    unsigned short* Wp3 = (unsigned short*)suballoc(9ul * 16 * 6 * 512 * 2);
    unsigned short* Wp4 = (unsigned short*)suballoc(9ul * 32 * 8 * 512 * 2);
    unsigned short* WlP = (unsigned short*)suballoc((size_t)8192 * 256 * 2);
    unsigned short* WsP = (unsigned short*)suballoc((size_t)4096 * 256 * 2);
    unsigned short* W0P = (unsigned short*)suballoc((size_t)512 * 512 * 2);
    // BN stats (scale, shift pairs)
    float* st_att = (float*)suballoc(12  * 2 * sizeof(float));
    float* st1    = (float*)suballoc(128 * 2 * sizeof(float));
    float* st2    = (float*)suballoc(192 * 2 * sizeof(float));
    float* st3    = (float*)suballoc(256 * 2 * sizeof(float));
    float* st4    = (float*)suballoc(512 * 2 * sizeof(float));
    // FC intermediates
    float*          o1   = (float*)suballoc((size_t)B * 256 * sizeof(float));
    float*          o2   = (float*)suballoc((size_t)B * 256 * sizeof(float));
    unsigned short* hcat = (unsigned short*)suballoc((size_t)B * 512 * 2);
    float*          h1   = (float*)suballoc((size_t)B * 512 * sizeof(float));
    float*          h2   = (float*)suballoc((size_t)B * 512 * sizeof(float));

    auto blocks = [](long total) -> unsigned int {
        return (unsigned int)((total + 255) / 256);
    };

    // --- weight packing (independent of activations) ---
    pack_conv_w<<<blocks(9l * 8  * 1 * 512), 256, 0, stream>>>(c1, Wp1, 128, 12);
    pack_conv_w<<<blocks(9l * 12 * 4 * 512), 256, 0, stream>>>(c2, Wp2, 192, 128);
    pack_conv_w<<<blocks(9l * 16 * 6 * 512), 256, 0, stream>>>(c3, Wp3, 256, 192);
    pack_conv_w<<<blocks(9l * 32 * 8 * 512), 256, 0, stream>>>(c4, Wp4, 512, 256);
    pack_gemm_w<<<blocks((long)8192 * 256), 256, 0, stream>>>(Wlast, WlP, 8192, 256, 16, 512);
    pack_gemm_w<<<blocks((long)4096 * 256), 256, 0, stream>>>(Wsec,  WsP, 4096, 256, 16, 256);
    pack_gemm_w<<<blocks((long)512  * 512), 256, 0, stream>>>(W0,    W0P, 512,  512, 0,  0);

    // --- front end + att BN (stats then in-place affine) ---
    frontend_att<<<B * 2, 256, 0, stream>>>(x, loc, ro, Wd, bd, Wl, bl, We, be, av, att);
    bn_stats_nhwc<<<12, 256, 0, stream>>>(att, g_in, b_in, st_att, 32, (long)B * 4761);
    bn_apply_nhwc<<<blocks((long)B * 4761 * 12), 256, 0, stream>>>(att, st_att, 12, 32, (long)B * 4761);

    // --- conv1: 12 -> 128, 69x69 pad1 ---
    conv_wmma_nhwc<<<dim3(5, 69, (unsigned)B * 2), 32, 0, stream>>>(
        att, Wp1, Y1, B, 12, 32, 69, 69, 128, 69, 69, 1);
    bn_stats_nhwc<<<128, 256, 0, stream>>>(Y1, g1, b1, st1, 128, (long)B * 4761);
    pool_nhwc<<<blocks((long)B * 34 * 34 * 16), 256, 0, stream>>>(
        Y1, st1, X2, B, 128, 69, 69, 34, 34, 0.2f, 1, 1);

    // --- conv2: 128 -> 192, 34x34 pad0 -> 32x32 ---
    conv_wmma_nhwc<<<dim3(2, 32, (unsigned)B * 3), 32, 0, stream>>>(
        X2, Wp2, Y2, B, 128, 128, 34, 34, 192, 32, 32, 0);
    bn_stats_nhwc<<<192, 256, 0, stream>>>(Y2, g2, b2, st2, 192, (long)B * 1024);
    pool_nhwc<<<blocks((long)B * 16 * 16 * 24), 256, 0, stream>>>(
        Y2, st2, X3, B, 192, 32, 32, 16, 16, 0.2f, 1, 1);

    // --- conv3: 192 -> 256, 16x16 pad1 ---
    conv_wmma_nhwc<<<dim3(1, 16, (unsigned)B * 4), 32, 0, stream>>>(
        X3, Wp3, Y3, B, 192, 192, 16, 16, 256, 16, 16, 1);
    bn_stats_nhwc<<<256, 256, 0, stream>>>(Y3, g3, b3, st3, 256, (long)B * 256);
    pool_nhwc<<<blocks((long)B * 8 * 8 * 32), 256, 0, stream>>>(
        Y3, st3, X4, B, 256, 16, 16, 8, 8, 0.2f, 1, 1);          // X4 = out3

    // --- conv4: 256 -> 512, 8x8 pad1 ---
    conv_wmma_nhwc<<<dim3(1, 8, (unsigned)B * 8), 32, 0, stream>>>(
        X4, Wp4, Y4, B, 256, 256, 8, 8, 512, 8, 8, 1);
    bn_stats_nhwc<<<512, 256, 0, stream>>>(Y4, g4, b4, st4, 512, (long)B * 64);
    pool_nhwc<<<blocks((long)B * 4 * 4 * 64), 256, 0, stream>>>(
        Y4, st4, X5, B, 512, 8, 8, 4, 4, 0.2f, 1, 1);            // X5 = out4 (f1)

    // --- f2 = maxpool(out3), no affine / no act ---
    pool_nhwc<<<blocks((long)B * 4 * 4 * 32), 256, 0, stream>>>(
        X4, nullptr, F2, B, 256, 8, 8, 4, 4, 0.0f, 0, 0);

    // --- FC heads + BN1d into bf16 concat ---
    gemm_wmma_packed<<<dim3(4, 8), 32, 0, stream>>>(X5, WlP, blast, o1, B, 256, 8192, 0);
    gemm_wmma_packed<<<dim3(4, 8), 32, 0, stream>>>(F2, WsP, bsec,  o2, B, 256, 4096, 0);
    bn1d<<<256, 128, 0, stream>>>(o1, glast, bglast, hcat, B, 256, 512, 0,   1);
    bn1d<<<256, 128, 0, stream>>>(o2, gsec,  bgsec,  hcat, B, 256, 512, 256, 1);

    // --- MLP: lrelu(hcat @ W0 + b0, 0.1) -> BN1d -> @ W1 + b1out ---
    gemm_wmma_packed<<<dim3(8, 8), 32, 0, stream>>>(hcat, W0P, b0, h1, B, 512, 512, 1);
    bn1d<<<512, 128, 0, stream>>>(h1, g0, bg0, h2, B, 512, 512, 0, 0);
    final_fc<<<(B * 5 + 127) / 128, 128, 0, stream>>>(h2, W1p, b1out, dout, B, 512, 5);
}
